// STGAT_63702954934297
// MI455X (gfx1250) — matrix operand order repaired and checked by reading, not compiled, not run on hardware
//
#include <hip/hip_runtime.h>

// Problem constants (match reference)
#define BB 2
#define TSZ 12
#define NNODE 5000
#define NEDGE 80000
#define HEADS 4
#define OUTD 16
#define EMB 64          // HEADS*OUTD
#define CCH 64
#define KW 3
#define HORIZ 12
#define NEG_SLOPE 0.2f

#define BTNH (BB*TSZ*NNODE*HEADS)   // 480000

typedef __attribute__((ext_vector_type(16))) __bf16          v16bf;
typedef __attribute__((ext_vector_type(16))) unsigned short  v16us;
typedef __attribute__((ext_vector_type(8)))  unsigned short  v8us;
typedef __attribute__((ext_vector_type(8)))  float           v8f;

__device__ __forceinline__ unsigned short f2bf(float f) {
    unsigned u = __float_as_uint(f);
    unsigned r = u + 0x7FFFu + ((u >> 16) & 1u);   // round-to-nearest-even
    return (unsigned short)(r >> 16);
}

__device__ __forceinline__ void atomicMaxF(float* a, float v) {
    // order-preserving int trick; works for mixed signs, init must be -inf
    if (v >= 0.f) atomicMax((int*)a, __float_as_int(v));
    else          atomicMin((unsigned int*)a, __float_as_uint(v));
}

// Load a 16-element bf16 fragment as two contiguous 16B LDS reads
__device__ __forceinline__ v16us ld_frag_2x8(const unsigned short* p) {
    v8us lo = *(const v8us*)(p);
    v8us hi = *(const v8us*)(p + 16);
    return __builtin_shufflevector(lo, hi, 0,1,2,3,4,5,6,7,8,9,10,11,12,13,14,15);
}

// ---------------- Stage 0: init workspace (m=-inf, den=num=0) ----------------
__global__ void STGAT_init(float* ws) {
    int i = blockIdx.x * blockDim.x + threadIdx.x;
    if (i < BTNH)            ws[i] = __int_as_float(0xFF800000);  // m = -inf
    else if (i < 3 * BTNH)   ws[i] = 0.f;                         // den, num
}

// ---------------- Stage 1: cs[h] = <Wg[h], a_src[h]>, cd likewise -----------
__global__ void STGAT_coeff(const float* __restrict__ Wg,
                            const float* __restrict__ a_src,
                            const float* __restrict__ a_dst,
                            float* __restrict__ coeff /* cs[4], cd[4] */) {
    int tid = threadIdx.x;
    if (tid < 8) {
        int h = tid & 3;
        const float* av = (tid < 4) ? a_src : a_dst;
        float acc = 0.f;
        for (int d = 0; d < OUTD; ++d) acc += Wg[h * OUTD + d] * av[h * OUTD + d];
        coeff[tid] = acc;
    }
}

// ---------------- Stage 2a: scatter-max of leaky-relu scores ----------------
__global__ void STGAT_edge_max(const float* __restrict__ x,
                               const int*   __restrict__ ei,
                               const float* __restrict__ coeff,
                               float* __restrict__ m) {
    int e  = blockIdx.x * blockDim.x + threadIdx.x;
    int bt = blockIdx.y;                     // 0..23 == b*TSZ + t
    if (e >= NEDGE) return;
    int src = ei[e];
    int dst = ei[NEDGE + e];
    float xs = x[bt * NNODE + src];
    float xd = x[bt * NNODE + dst];
    int base = (bt * NNODE + dst) * HEADS;
    #pragma unroll
    for (int h = 0; h < HEADS; ++h) {
        float sc = xs * coeff[h] + xd * coeff[4 + h];
        sc = (sc > 0.f) ? sc : NEG_SLOPE * sc;
        atomicMaxF(&m[base + h], sc);
    }
}

// ---------------- Stage 2b: exp-sum and weighted-source sum -----------------
__global__ void STGAT_edge_sum(const float* __restrict__ x,
                               const int*   __restrict__ ei,
                               const float* __restrict__ coeff,
                               const float* __restrict__ m,
                               float* __restrict__ den,
                               float* __restrict__ num) {
    int e  = blockIdx.x * blockDim.x + threadIdx.x;
    int bt = blockIdx.y;
    if (e >= NEDGE) return;
    int src = ei[e];
    int dst = ei[NEDGE + e];
    float xs = x[bt * NNODE + src];
    float xd = x[bt * NNODE + dst];
    int base = (bt * NNODE + dst) * HEADS;
    #pragma unroll
    for (int h = 0; h < HEADS; ++h) {
        float sc = xs * coeff[h] + xd * coeff[4 + h];
        sc = (sc > 0.f) ? sc : NEG_SLOPE * sc;
        float ex = __expf(sc - m[base + h]);     // m finite for non-empty segments
        atomicAdd(&den[base + h], ex);
        atomicAdd(&num[base + h], ex * xs);
    }
}

// ------------- Stage 3: s[b][n][h][t] = num/(den+1e-16) (re-layout) ---------
__global__ void STGAT_build_s(const float* __restrict__ den,
                              const float* __restrict__ num,
                              float* __restrict__ s) {
    int i = blockIdx.x * blockDim.x + threadIdx.x;
    if (i >= BTNH) return;
    int t = i % TSZ;
    int r = i / TSZ;
    int h = r & 3;
    int r2 = r >> 2;
    int n = r2 % NNODE;
    int b = r2 / NNODE;
    int src_idx = ((b * TSZ + t) * NNODE + n) * HEADS + h;
    s[i] = num[src_idx] / (den[src_idx] + 1e-16f);
}

// ------------- Stage 4: im2col conv GEMM + relu/mean + head GEMM (WMMA) -----
// grid: (ceil(N/16), B), block: 128 (4 waves). Wave w owns out-channels [16w,16w+16).
// Contraction dim K = k*64 + c (192), chunk ck covers K in [32ck, 32ck+32):
//   k = ck>>1, c = (ck&1)*32 + Klocal.
// Time axis in LDS is zero-padded: physical row tp = t+1, tp=0 and tp=13 are zeros,
// so the inner loop is branch-free.
__global__ void STGAT_conv_head(const float* __restrict__ s,     // [B][N][H][T]
                                const float* __restrict__ Wg,    // [4][16]
                                const float* __restrict__ bg,    // [64]
                                const float* __restrict__ Wc,    // [64][64][3]
                                const float* __restrict__ bc,    // [64]
                                const float* __restrict__ Wh,    // [12][64]
                                const float* __restrict__ bh,    // [12]
                                float* __restrict__ out) {       // [B][12][N]
    // Fragment-native LDS layouts (wide ds_load friendly)
    __shared__ __align__(32) unsigned short Awc_s[CCH * 6 * 32];       // [o][ck][Klocal]
    __shared__ __align__(32) unsigned short Ht2_s[(TSZ + 2) * 16 * EMB]; // [tp][node][c]
    __shared__ __align__(32) unsigned short Wha_s[16 * 2 * 32];        // [row][ck2][Klocal]
    __shared__ __align__(32) unsigned short zbT_s[16 * EMB];           // [node][o]
    __shared__ float bc_s[EMB], bh_s[HORIZ];

    const int tid  = threadIdx.x;
    const int lane = tid & 31;
    const int w    = tid >> 5;          // wave id 0..3
    const int oL   = lane & 15;
    const int hi   = lane >> 4;         // lane half
    const int col  = lane & 15;         // output column (node / N-dim)
    const int n0   = blockIdx.x * 16;
    const int b    = blockIdx.y;

    // ---- cooperative staging (one-time) ----
    // A operand: Awc_s[(o*6+ck)*32 + K] = Wc[o][ (ck&1)*32+K ][ ck>>1 ]
    for (int i = tid; i < CCH * 6 * 32; i += 128) {
        int K  = i & 31;
        int ck = (i >> 5) % 6;
        int o  = i / 192;
        int c  = ((ck & 1) << 5) + K;
        Awc_s[i] = f2bf(Wc[(o * EMB + c) * KW + (ck >> 1)]);
    }
    // Head A operand, zero-padded rows 12..15
    for (int i = tid; i < 16 * 2 * 32; i += 128) {
        int K = i & 31;
        int ck2 = (i >> 5) & 1;
        int row = i >> 6;
        Wha_s[i] = (row < HORIZ) ? f2bf(Wh[row * EMB + (ck2 << 5) + K])
                                 : (unsigned short)0;
    }
    if (tid < EMB)   bc_s[tid] = bc[tid];
    if (tid < HORIZ) bh_s[tid] = bh[tid];
    // B operand source with zero-padded time boundary rows:
    //   Ht2_s[tp][node][c] = (1<=tp<=12) ? s[b][n][c>>4][tp-1]*Wg[c] + bg[c] : 0
    for (int i = tid; i < (TSZ + 2) * 16 * EMB; i += 128) {
        int c  = i & 63;
        int nl = (i >> 6) & 15;
        int tp = i >> 10;
        int t  = tp - 1;
        int n  = n0 + nl;
        unsigned short v = 0;
        if (t >= 0 && t < TSZ) {
            float sv = (n < NNODE) ? s[((b * NNODE + n) * HEADS + (c >> 4)) * TSZ + t] : 0.f;
            v = f2bf(sv * Wg[c] + bg[c]);
        }
        Ht2_s[i] = v;
    }
    __syncthreads();

    // ---- A fragments resident in VGPRs (48 VGPRs) ----
    const int o = (w << 4) + oL;
    v16us afrag[6];
    #pragma unroll
    for (int ck = 0; ck < 6; ++ck)
        afrag[ck] = *(const v16us*)&Awc_s[((o * 6 + ck) << 5) + (hi << 4)];

    v8f acc = {};
    #pragma clang loop unroll(disable)
    for (int t = 0; t < TSZ; ++t) {
        // two independent accumulation chains for XDL ILP; branch-free B loads
        v8f z0 = {}, z1 = {};
        #pragma unroll
        for (int ck = 0; ck < 6; ++ck) {
            int tp = t + (ck >> 1);               // physical row = t + k (padded)
            v16us bfrag = *(const v16us*)&Ht2_s[((tp * 16 + col) << 6) +
                                                ((ck & 1) << 5) + (hi << 4)];
            if (ck & 1)
                z1 = __builtin_amdgcn_wmma_f32_16x16x32_bf16(
                        false, __builtin_bit_cast(v16bf, afrag[ck]),
                        false, __builtin_bit_cast(v16bf, bfrag),
                        (short)0, z1, false, false);
            else
                z0 = __builtin_amdgcn_wmma_f32_16x16x32_bf16(
                        false, __builtin_bit_cast(v16bf, afrag[ck]),
                        false, __builtin_bit_cast(v16bf, bfrag),
                        (short)0, z0, false, false);
        }
        #pragma unroll
        for (int r = 0; r < 8; ++r) {
            int oo = (w << 4) + r + (hi << 3);
            acc[r] += fmaxf(z0[r] + z1[r] + bc_s[oo], 0.f) * (1.f / 12.f);
        }
    }

    // ---- spill zbar tile to LDS transposed: zbT_s[node][o] ----
    #pragma unroll
    for (int r = 0; r < 8; ++r) {
        int oo = (w << 4) + r + (hi << 3);
        zbT_s[(col << 6) + oo] = f2bf(acc[r]);
    }
    __syncthreads();

    // ---- head GEMM: y[12x16] = Wh_pad[16x64] x zbar[64x16], wave 0 only ----
    if (w == 0) {
        v8f y = {};
        #pragma unroll
        for (int ck2 = 0; ck2 < 2; ++ck2) {
            v16us wa = *(const v16us*)&Wha_s[((oL * 2 + ck2) << 5) + (hi << 4)];
            v16us wb = *(const v16us*)&zbT_s[(col << 6) + (ck2 << 5) + (hi << 4)];
            y = __builtin_amdgcn_wmma_f32_16x16x32_bf16(
                    false, __builtin_bit_cast(v16bf, wa),
                    false, __builtin_bit_cast(v16bf, wb),
                    (short)0, y, false, false);
        }
        int n = n0 + col;
        #pragma unroll
        for (int r = 0; r < 8; ++r) {
            int hor = r + (hi << 3);
            if (hor < HORIZ && n < NNODE)
                out[(b * HORIZ + hor) * NNODE + n] = y[r] + bh_s[hor];
        }
    }
}

extern "C" void kernel_launch(void* const* d_in, const int* in_sizes, int n_in,
                              void* d_out, int out_size, void* d_ws, size_t ws_size,
                              hipStream_t stream) {
    const float* x     = (const float*)d_in[0];
    const int*   ei    = (const int*)  d_in[1];
    const float* Wg    = (const float*)d_in[2];
    const float* a_src = (const float*)d_in[3];
    const float* a_dst = (const float*)d_in[4];
    const float* bg    = (const float*)d_in[5];
    const float* Wc    = (const float*)d_in[6];
    const float* bc    = (const float*)d_in[7];
    const float* Wh    = (const float*)d_in[8];
    const float* bh    = (const float*)d_in[9];
    float* out = (float*)d_out;

    float* ws    = (float*)d_ws;
    float* m     = ws;                 // [24][N][4]
    float* den   = ws + BTNH;
    float* num   = ws + 2 * BTNH;
    float* sarr  = ws + 3 * BTNH;      // [B][N][4][12]
    float* coeff = ws + 4 * BTNH;      // cs[4], cd[4]

    STGAT_init<<<(3 * BTNH + 255) / 256, 256, 0, stream>>>(ws);
    STGAT_coeff<<<1, 32, 0, stream>>>(Wg, a_src, a_dst, coeff);

    dim3 egrid((NEDGE + 255) / 256, BB * TSZ);
    STGAT_edge_max<<<egrid, 256, 0, stream>>>(x, ei, coeff, m);
    STGAT_edge_sum<<<egrid, 256, 0, stream>>>(x, ei, coeff, m, den, num);

    STGAT_build_s<<<(BTNH + 255) / 256, 256, 0, stream>>>(den, num, sarr);

    dim3 cgrid((NNODE + 15) / 16, BB);
    STGAT_conv_head<<<cgrid, 128, 0, stream>>>(sarr, Wg, bg, Wc, bc, Wh, bh, out);
}
